// Head_71373766525017
// MI455X (gfx1250) — matrix operand order repaired and checked by reading, compiled, not run on hardware
//
#include <hip/hip_runtime.h>

// ---------------------------------------------------------------------------
// Head forward (single attention head), B=8 T=2048 C=1024 H=128, fp32 I/O.
// bf16 WMMA (v_wmma_f32_16x16x32_bf16), fp32 accumulation.
//   Kernel 0a: x fp32 -> bf16 [BT][C]        (memory-bound, one pass)
//   Kernel 0b: W fp32 -> bf16 transposed [H][C] (x3, tiny, L2-resident after)
//   Kernel 1 : fused q/k/v projection, pure load+WMMA; each wave computes
//              32 rows so every B fragment feeds two back-to-back WMMAs
//   Kernel 2 : flash attention, double-buffered GLOBAL_LOAD_ASYNC_TO_LDS_B128
//              staging of K/V tiles (ASYNCcnt overlap), fp32 out [B][T][H]
// ---------------------------------------------------------------------------

#define BB 8
#define TT 2048
#define CC 1024
#define HH 128
#define BT (BB * TT)

typedef __attribute__((ext_vector_type(16))) __bf16 bf16x16;
typedef __attribute__((ext_vector_type(8)))  __bf16 bf16x8;
typedef __attribute__((ext_vector_type(8)))  float  f32x8;

// exact pointee type the async-to-LDS builtin wants
typedef __attribute__((__vector_size__(16))) int i32x4v;
typedef __attribute__((address_space(1))) i32x4v as1_i32x4;
typedef __attribute__((address_space(3))) i32x4v as3_i32x4;

#if defined(__has_builtin)
#if __has_builtin(__builtin_amdgcn_global_load_async_to_lds_b128)
#define HAVE_ASYNC_LDS 1
#endif
#if __has_builtin(__builtin_amdgcn_perm)
#define HAVE_PERM 1
#endif
#endif

__device__ __forceinline__ unsigned rne_u(float f) {      // bf16 bits in [31:16]
    unsigned u = __builtin_bit_cast(unsigned, f);
    return u + 0x7FFFu + ((u >> 16) & 1u);
}

__device__ __forceinline__ unsigned short f2bfu(float f) {
    return (unsigned short)(rne_u(f) >> 16);
}

// pack two fp32 -> two bf16 in one dword (a -> low, b -> high); v_perm_b32 pack
__device__ __forceinline__ unsigned pkbf2(float a, float b) {
    unsigned ua = rne_u(a), ub = rne_u(b);
#if HAVE_PERM
    return __builtin_amdgcn_perm(ub, ua, 0x07060302u);
#else
    return (ua >> 16) | (ub & 0xFFFF0000u);
#endif
}

__device__ __forceinline__ bf16x16 cat16(bf16x8 lo, bf16x8 hi) {
    return __builtin_shufflevector(lo, hi, 0,1,2,3,4,5,6,7,8,9,10,11,12,13,14,15);
}

// two 8-element (16B) runs at p and p+elem_off
__device__ __forceinline__ bf16x16 ld2x8(const unsigned short* p, int elem_off) {
    bf16x8 lo = *(const bf16x8*)p;
    bf16x8 hi = *(const bf16x8*)(p + elem_off);
    return cat16(lo, hi);
}

__device__ __forceinline__ f32x8 wmma_bf16(bf16x16 a, bf16x16 b, f32x8 c) {
    return __builtin_amdgcn_wmma_f32_16x16x32_bf16(false, a, false, b,
                                                   (short)0, c, false, false);
}

// 16B global -> LDS copy: async (ASYNCcnt) when available, sync fallback
__device__ __forceinline__ void cp16_g2l(const unsigned short* g, unsigned short* l) {
#if HAVE_ASYNC_LDS
    __builtin_amdgcn_global_load_async_to_lds_b128(
        (as1_i32x4*)(unsigned long long)g,
        (as3_i32x4*)(unsigned int)(unsigned long long)l,
        0, 0);
#else
    uint4 t = *(const uint4*)g;
    *(uint4*)l = t;
#endif
}

__device__ __forceinline__ void wait_async(int n) {
#if HAVE_ASYNC_LDS
    if (n == 0) asm volatile("s_wait_asynccnt 0x0" ::: "memory");
    else        asm volatile("s_wait_asynccnt 0x8" ::: "memory");
#endif
}

// ---------------------------------------------------------------------------
// Kernel 0a: x fp32 -> bf16, 8 elements/thread.  grid = BT*CC/(256*8)
// ---------------------------------------------------------------------------
__global__ __launch_bounds__(256)
void cvt_x(const float* __restrict__ x, unsigned short* __restrict__ xbf) {
    size_t base = ((size_t)blockIdx.x * 256 + threadIdx.x) * 8;
    float4 a = *(const float4*)(x + base);
    float4 b = *(const float4*)(x + base + 4);
    uint4 o;
    o.x = pkbf2(a.x, a.y); o.y = pkbf2(a.z, a.w);
    o.z = pkbf2(b.x, b.y); o.w = pkbf2(b.z, b.w);
    *(uint4*)(xbf + base) = o;
}

// ---------------------------------------------------------------------------
// Kernel 0b: W [C][H] fp32 -> Wt [H][C] bf16.  grid = (CC/64, 3)
// ---------------------------------------------------------------------------
__global__ __launch_bounds__(256)
void cvt_w(const float* __restrict__ Wq, const float* __restrict__ Wk,
           const float* __restrict__ Wv,
           unsigned short* __restrict__ WtQ, unsigned short* __restrict__ WtK,
           unsigned short* __restrict__ WtV) {
    const int which = blockIdx.y;
    const float* W = (which == 0) ? Wq : (which == 1) ? Wk : Wv;
    unsigned short* Wt = (which == 0) ? WtQ : (which == 1) ? WtK : WtV;
    const int k0 = blockIdx.x * 64;
    #pragma unroll
    for (int i = 0; i < 32; ++i) {
        int idx = threadIdx.x + 256 * i;       // 0..8191 over a 64x128 chunk
        int r = idx >> 7, c = idx & 127;
        Wt[(size_t)c * CC + k0 + r] = f2bfu(W[(size_t)(k0 + r) * HH + c]);
    }
}

// ---------------------------------------------------------------------------
// Kernel 1: y = x @ W, all-bf16 inputs, pure load+WMMA.
// grid = (BT/256, 3); block = 256 (8 waves); each wave: 32 rows x 128 cols,
// so each B fragment is reused by two independent WMMAs (no RAW hazard).
// which: 0 -> q [T][H], 1 -> k [T][H], 2 -> v transposed [H][T]
// ---------------------------------------------------------------------------
__global__ __launch_bounds__(256)
void qkv_proj(const unsigned short* __restrict__ xbf,
              const unsigned short* __restrict__ WtQ,
              const unsigned short* __restrict__ WtK,
              const unsigned short* __restrict__ WtV,
              unsigned short* __restrict__ qbf, unsigned short* __restrict__ kbf,
              unsigned short* __restrict__ vt) {
    const int tid  = threadIdx.x;
    const int w    = tid >> 5;
    const int lane = tid & 31;
    const int half = lane >> 4;
    const int n15  = lane & 15;

    const int which = blockIdx.y;
    const unsigned short* Wt = (which == 0) ? WtQ : (which == 1) ? WtK : WtV;

    const int row0  = blockIdx.x * 256;
    const int arow0 = row0 + w * 16 + n15;          // A rows, group 0
    const int arow1 = row0 + 128 + w * 16 + n15;    // A rows, group 1

    f32x8 acc0[8] = {}, acc1[8] = {};

    for (int k0 = 0; k0 < CC; k0 += 32) {
        // A fragments: xbf[row][k0 + {8h..8h+7, 16+8h..16+8h+7}] (2 x b128 each)
        bf16x16 A0 = ld2x8(xbf + (size_t)arow0 * CC + k0 + 8 * half, 16);
        bf16x16 A1 = ld2x8(xbf + (size_t)arow1 * CC + k0 + 8 * half, 16);
        #pragma unroll
        for (int f = 0; f < 8; ++f) {
            // B fragment: Wt[f*16+n15][k0 + {8h..8h+7, 16+8h..16+8h+7}] (2 x b128)
            bf16x16 Bm = ld2x8(Wt + (size_t)(f * 16 + n15) * CC + k0 + 8 * half, 16);
            acc0[f] = wmma_bf16(A0, Bm, acc0[f]);   // same B, two independent
            acc1[f] = wmma_bf16(A1, Bm, acc1[f]);   // accumulators
        }
    }

    #pragma unroll
    for (int g = 0; g < 2; ++g) {
        const int rbase = row0 + g * 128 + w * 16;
        if (which != 2) {
            unsigned short* dst = (which == 0) ? qbf : kbf;
            #pragma unroll
            for (int f = 0; f < 8; ++f)
                #pragma unroll
                for (int r = 0; r < 8; ++r) {
                    int rr = rbase + r + 8 * half;
                    float v = g ? acc1[f][r] : acc0[f][r];
                    dst[(size_t)rr * HH + f * 16 + n15] = f2bfu(v);
                }
        } else {
            // v^T: 8 accumulator rows per lane are contiguous t -> one b128 store
            int b  = row0 / TT;                      // 256 | T so same batch
            int tB = (rbase % TT) + 8 * half;
            #pragma unroll
            for (int f = 0; f < 8; ++f) {
                uint4 o;
                if (g == 0) {
                    o.x = pkbf2(acc0[f][0], acc0[f][1]);
                    o.y = pkbf2(acc0[f][2], acc0[f][3]);
                    o.z = pkbf2(acc0[f][4], acc0[f][5]);
                    o.w = pkbf2(acc0[f][6], acc0[f][7]);
                } else {
                    o.x = pkbf2(acc1[f][0], acc1[f][1]);
                    o.y = pkbf2(acc1[f][2], acc1[f][3]);
                    o.z = pkbf2(acc1[f][4], acc1[f][5]);
                    o.w = pkbf2(acc1[f][6], acc1[f][7]);
                }
                int h = f * 16 + n15;
                *(uint4*)(vt + ((size_t)b * HH + h) * TT + tB) = o;
            }
        }
    }
}

// ---------------------------------------------------------------------------
// Kernel 2: flash attention.  grid = B*T/64; block = 128 (4 waves, 64 q-rows).
// Double-buffered async staging of K (32x128) and V^T (128x32) tiles in LDS.
// ---------------------------------------------------------------------------
__global__ __launch_bounds__(128)
void flash_attn(const unsigned short* __restrict__ qbf,
                const unsigned short* __restrict__ kbf,
                const unsigned short* __restrict__ vt,
                float* __restrict__ out) {
    __shared__ __align__(16) unsigned short Kt[2][32][136];
    __shared__ __align__(16) unsigned short Vs[2][128][40];
    __shared__ __align__(16) unsigned short pl[4][16][40];   // per-wave P tile

    const int tid  = threadIdx.x;
    const int w    = tid >> 5;
    const int lane = tid & 31;
    const int half = lane >> 4;
    const int n15  = lane & 15;

    const int blocksPerB = TT / 64;
    const int b  = blockIdx.x / blocksPerB;
    const int t0 = (blockIdx.x % blocksPerB) * 64;
    const int q0 = t0 + w * 16;

    const unsigned short* qB = qbf + (size_t)b * TT * HH;
    const unsigned short* kB = kbf + (size_t)b * TT * HH;
    const unsigned short* vB = vt  + (size_t)b * HH * TT;

    // Q fragments (A layout), resident for the whole loop
    bf16x16 Qf[4];
    #pragma unroll
    for (int hc = 0; hc < 4; ++hc) {
        const unsigned short* p = qB + (size_t)(q0 + n15) * HH + hc * 32 + 8 * half;
        Qf[hc] = ld2x8(p, 16);
    }

    f32x8 O[8] = {};
    float mrow[8], lrow[8];
    #pragma unroll
    for (int r = 0; r < 8; ++r) { mrow[r] = -3.0e38f; lrow[r] = 0.0f; }

    const int nsteps = ((t0 + 63) >> 5) + 1;   // uniform across the block
    const float sc = 0.03125f;                 // C^-0.5 = 1/32

    // stage step s into buffer buf (8 async b128 per thread)
    auto stage = [&](int sb, int buf) {
        #pragma unroll
        for (int i = 0; i < 4; ++i) {
            int c  = tid + 128 * i;                    // 0..511 16B chunks
            int kr = c >> 3, kc = (c & 7) * 8;
            cp16_g2l(kB + (size_t)(sb + kr) * HH + kc, &Kt[buf][kr][kc]);
            int vh = c >> 2, vc = (c & 3) * 8;
            cp16_g2l(vB + (size_t)vh * TT + sb + vc, &Vs[buf][vh][vc]);
        }
    };

    stage(0, 0);                                       // prologue

    for (int s = 0; s < nsteps; ++s) {
        const int sb  = s * 32;
        const int cur = s & 1;

        asm volatile("s_wait_dscnt 0x0" ::: "memory"); // our LDS reads landed
        __syncthreads();                               // all done with buf cur^1
        if (s + 1 < nsteps) {                          // prefetch next tile
            stage((s + 1) * 32, cur ^ 1);
            wait_async(8);                             // stage(s) complete
        } else {
            wait_async(0);
        }
        __syncthreads();                               // buf cur visible

        // ---- S = Q K^T : two 16x16 tiles over 32 keys --------------------
        f32x8 S0 = {}, S1 = {};
        #pragma unroll
        for (int hc = 0; hc < 4; ++hc) {
            bf16x16 B0 = ld2x8(&Kt[cur][n15][hc * 32 + 8 * half], 16);
            bf16x16 B1 = ld2x8(&Kt[cur][16 + n15][hc * 32 + 8 * half], 16);
            S0 = wmma_bf16(Qf[hc], B0, S0);
            S1 = wmma_bf16(Qf[hc], B1, S1);
        }

        // ---- scale, causal mask, online softmax --------------------------
        float nm[8], alpha[8];
        #pragma unroll
        for (int r = 0; r < 8; ++r) {
            int qrow = q0 + r + 8 * half;
            float s0 = S0[r] * sc, s1 = S1[r] * sc;
            if (sb + n15 > qrow)      s0 = -3.0e38f;
            if (sb + 16 + n15 > qrow) s1 = -3.0e38f;
            S0[r] = s0; S1[r] = s1;
            float mx = fmaxf(s0, s1);
            mx = fmaxf(mx, __shfl_xor(mx, 1, 16));
            mx = fmaxf(mx, __shfl_xor(mx, 2, 16));
            mx = fmaxf(mx, __shfl_xor(mx, 4, 16));
            mx = fmaxf(mx, __shfl_xor(mx, 8, 16));
            nm[r] = fmaxf(mrow[r], mx);
        }
        #pragma unroll
        for (int r = 0; r < 8; ++r) {
            alpha[r] = __expf(mrow[r] - nm[r]);
            float p0 = __expf(S0[r] - nm[r]);
            float p1 = __expf(S1[r] - nm[r]);
            S0[r] = p0; S1[r] = p1;
            float rs = p0 + p1;
            rs += __shfl_xor(rs, 1, 16);
            rs += __shfl_xor(rs, 2, 16);
            rs += __shfl_xor(rs, 4, 16);
            rs += __shfl_xor(rs, 8, 16);
            lrow[r] = lrow[r] * alpha[r] + rs;
            mrow[r] = nm[r];
        }
        #pragma unroll
        for (int f = 0; f < 8; ++f)
            #pragma unroll
            for (int r = 0; r < 8; ++r)
                O[f][r] *= alpha[r];

        // ---- C-layout -> A-layout for P via per-wave LDS tile ------------
        #pragma unroll
        for (int r = 0; r < 8; ++r) {
            unsigned e = pkbf2(S0[r], S1[r]);
            pl[w][r + 8 * half][n15]      = (unsigned short)e;
            pl[w][r + 8 * half][16 + n15] = (unsigned short)(e >> 16);
        }
        asm volatile("s_wait_dscnt 0x0" ::: "memory");   // wave-local RAW fence
        bf16x16 Pf = ld2x8(&pl[w][n15][8 * half], 16);

        // ---- O += P V ----------------------------------------------------
        #pragma unroll
        for (int f = 0; f < 8; ++f) {
            bf16x16 Vf = ld2x8(&Vs[cur][f * 16 + n15][8 * half], 16);
            O[f] = wmma_bf16(Pf, Vf, O[f]);
        }
    }

    // ---- epilogue: divide by row sums, fp32 store ------------------------
    #pragma unroll
    for (int f = 0; f < 8; ++f) {
        #pragma unroll
        for (int r = 0; r < 8; ++r) {
            int qrow = q0 + r + 8 * half;
            out[((size_t)b * TT + qrow) * HH + f * 16 + n15] = O[f][r] / lrow[r];
        }
    }
}

// ---------------------------------------------------------------------------
extern "C" void kernel_launch(void* const* d_in, const int* in_sizes, int n_in,
                              void* d_out, int out_size, void* d_ws, size_t ws_size,
                              hipStream_t stream) {
    const float* x  = (const float*)d_in[0];
    const float* Wk = (const float*)d_in[1];
    const float* Wq = (const float*)d_in[2];
    const float* Wv = (const float*)d_in[3];
    float* out = (float*)d_out;

    // workspace layout (bf16): q | k | v^T | xbf | WtQ | WtK | WtV
    const size_t seg  = (size_t)BB * TT * HH * 2;         // 4 MB
    const size_t xsz  = (size_t)BT * CC * 2;              // 32 MB
    const size_t wsz  = (size_t)HH * CC * 2;              // 256 KB
    char* p = (char*)d_ws;
    unsigned short* qbf = (unsigned short*)p;             p += seg;
    unsigned short* kbf = (unsigned short*)p;             p += seg;
    unsigned short* vt  = (unsigned short*)p;             p += seg;
    unsigned short* xbf = (unsigned short*)p;             p += xsz;
    unsigned short* WtQ = (unsigned short*)p;             p += wsz;
    unsigned short* WtK = (unsigned short*)p;             p += wsz;
    unsigned short* WtV = (unsigned short*)p;

    cvt_x<<<(BT * (size_t)CC) / (256 * 8), 256, 0, stream>>>(x, xbf);
    cvt_w<<<dim3(CC / 64, 3, 1), 256, 0, stream>>>(Wq, Wk, Wv, WtQ, WtK, WtV);
    qkv_proj<<<dim3(BT / 256, 3, 1), 256, 0, stream>>>(xbf, WtQ, WtK, WtV,
                                                       qbf, kbf, vt);
    flash_attn<<<BB * TT / 64, 128, 0, stream>>>(qbf, kbf, vt, out);
}
